// BiFuser_N_1924145349146
// MI455X (gfx1250) — matrix operand order repaired and checked by reading, not compiled, not run on hardware
//
#include <hip/hip_runtime.h>
#include <hip/hip_bf16.h>

typedef __attribute__((ext_vector_type(16))) _Float16 v16h;
typedef __attribute__((ext_vector_type(8)))  _Float16 v8h;
typedef __attribute__((ext_vector_type(8)))  float    v8f;
typedef int v4i __attribute__((vector_size(16)));   // pointee type the async builtin expects

#define S_VOX (96*96*16)   /* 147456 voxels */
#define NPT   2000

// ---- async global->LDS path (gfx1250), with sync fallback for host pass / older clang ----
#if defined(__gfx1250__) && __has_builtin(__builtin_amdgcn_global_load_async_to_lds_b128) && \
    __has_builtin(__builtin_amdgcn_s_wait_asynccnt)
#define HAVE_ASYNC_LDS 1
#else
#define HAVE_ASYNC_LDS 0
#endif

__device__ __forceinline__ void cp_g2l_b128(const _Float16* g, _Float16* l) {
#if HAVE_ASYNC_LDS
  __builtin_amdgcn_global_load_async_to_lds_b128(
      (__attribute__((address_space(1))) v4i*)g,
      (__attribute__((address_space(3))) v4i*)l, 0, 0);
#else
  *(v8h*)l = *(const v8h*)g;
#endif
}

__device__ __forceinline__ void async_fence() {
#if HAVE_ASYNC_LDS
  __builtin_amdgcn_s_wait_asynccnt(0);
#endif
}

// ---------- WMMA fragment helpers (wave32; A and B both use contiguous k = 16*(lane/16) + s) ----------

__device__ __forceinline__ v16h frag16(const _Float16* p0, const _Float16* p1) {
  v8h lo = *(const v8h*)p0;
  v8h hi = *(const v8h*)p1;
  return __builtin_shufflevector(lo, hi, 0,1,2,3,4,5,6,7,8,9,10,11,12,13,14,15);
}

// A (16xK tile, row-major in LDS, stride lda halfs):
// lane<16: row=lane, elems = K 0..15 ; lane>=16: row=lane-16, elems = K 16..31
__device__ __forceinline__ v16h load_a_frag(const _Float16* sA, int lda) {
  int lane = threadIdx.x & 31;
  int row  = lane & 15;
  int kb   = (lane >> 4) * 16;
  const _Float16* base = sA + row * lda + kb;
  return frag16(base, base + 8);
}

// B (32x16 tile stored N-major/K-contiguous: sBt[n][k], stride ldb halfs):
// lane<16: col=lane, elems = K 0..15 ; lane>=16: col=lane-16, elems = K 16..31
__device__ __forceinline__ v16h load_b_frag(const _Float16* sBt, int ldb) {
  int lane = threadIdx.x & 31;
  int col  = lane & 15;
  int kb   = (lane >> 4) * 16;
  const _Float16* base = sBt + col * ldb + kb;
  return frag16(base, base + 8);
}

__device__ __forceinline__ v8f wmma_f16(v16h a, v16h b, v8f c) {
  return __builtin_amdgcn_wmma_f32_16x16x32_f16(false, a, false, b, (short)0, c, false, false);
}

// ---------- KNN: one block per query, 256-thread strided min-reduce ----------

__global__ __launch_bounds__(256) void knn_kernel(
    const int* __restrict__ iq, const int* __restrict__ ik, int nk, int* __restrict__ nn)
{
  __shared__ float sd[256];
  __shared__ int   si[256];
  const int q = blockIdx.x, tid = threadIdx.x;
  const float qh = (float)iq[q*4+1], qw = (float)iq[q*4+2], ql = (float)iq[q*4+3];
  float bd = 3.4e38f; int bi = 0;
  for (int k = tid; k < nk; k += 256) {
    float dh = qh - (float)ik[k*4+1];
    float dw = qw - (float)ik[k*4+2];
    float dl = ql - (float)ik[k*4+3];
    float d2 = dh*dh + dw*dw + dl*dl;
    if (d2 < bd) { bd = d2; bi = k; }
  }
  sd[tid] = bd; si[tid] = bi;
  __syncthreads();
  for (int s = 128; s > 0; s >>= 1) {
    if (tid < s && sd[tid+s] < sd[tid]) { sd[tid] = sd[tid+s]; si[tid] = si[tid+s]; }
    __syncthreads();
  }
  if (tid == 0) nn[q] = (sd[0] < 13.3f*13.3f) ? si[0] : -1;
}

// ---------- gather -> relu(x @ lin_w^T + b) * sel -> scatter into fused f16 slab ----------
// One block = 16 queries x 128 out channels; 8 waves, one 16x16 WMMA tile each.

__global__ __launch_bounds__(256) void fuse_knn_linear(
    const float* __restrict__ grid_src, const float* __restrict__ grid_sel,
    const int* __restrict__ inds_src, const int* __restrict__ inds_q,
    const int* __restrict__ nn, const _Float16* __restrict__ lwh,
    const float* __restrict__ lin_b, _Float16* __restrict__ X1, int fused_off)
{
  constexpr int LDA = 40;
  __shared__ __align__(16) _Float16 sA[16 * LDA];
  const int tid = threadIdx.x, lane = tid & 31, wave = tid >> 5;
  const int qbase = blockIdx.x * 16;
  const int r = tid >> 4, cp = (tid & 15) * 2;

  // spatial offset of the nearest-neighbor source point for row r (JAX -1 wraps to last)
  int idx = nn[qbase + r];
  if (idx < 0) idx = NPT - 1;
  const long ssrc = (long)((inds_src[idx*4+1] * 96 + inds_src[idx*4+2]) * 16 + inds_src[idx*4+3]);

  const int colg = wave * 16 + (lane & 15);
  v8f acc = {};
  for (int kc = 0; kc < 128; kc += 32) {
    sA[r * LDA + cp    ] = (_Float16)grid_src[(long)(kc + cp    ) * S_VOX + ssrc];
    sA[r * LDA + cp + 1] = (_Float16)grid_src[(long)(kc + cp + 1) * S_VOX + ssrc];
    __syncthreads();
    v16h a = load_a_frag(sA, LDA);
    const _Float16* bp = lwh + (long)colg * 128 + kc + (lane >> 4) * 16;
    v16h b = frag16(bp, bp + 8);
    acc = wmma_f16(a, b, acc);
    __syncthreads();
  }
  const int crow = (lane >> 4) * 8;
  const float bias = lin_b[colg];
  for (int rr = 0; rr < 8; ++rr) {
    const int qq = qbase + crow + rr;
    const long sq = (long)((inds_q[qq*4+1] * 96 + inds_q[qq*4+2]) * 16 + inds_q[qq*4+3]);
    const float sel = grid_sel[(long)colg * S_VOX + sq];
    const float v = fmaxf(acc[rr] + bias, 0.f) * sel;
    X1[sq * 512 + fused_off + colg] = (_Float16)v;
  }
}

// ---------- pack NCHWL f32 -> NHWC f16 slab [s][512], zero fused sections ----------

__global__ __launch_bounds__(256) void pack_inputs(
    const float* __restrict__ img, const float* __restrict__ pts, _Float16* __restrict__ X1)
{
  const long idx = (long)blockIdx.x * 256 + threadIdx.x;   // over S_VOX*128
  const int  c = (int)(idx & 127);
  const long s = idx >> 7;
  _Float16* row = X1 + s * 512;
  row[c]       = (_Float16)img[(long)c * S_VOX + s];
  row[128 + c] = (_Float16)pts[(long)c * S_VOX + s];
  row[256 + c] = (_Float16)0.f;
  row[384 + c] = (_Float16)0.f;
}

// ---------- weight repack: [COUT][CIN][3][3][3] f32 -> [COUT][tap][CIN] f16 ----------

template<int CIN, int COUT>
__global__ __launch_bounds__(256) void prep_conv_w(
    const float* __restrict__ Ws, _Float16* __restrict__ Wt)
{
  const long idx = (long)blockIdx.x * 256 + threadIdx.x;
  if (idx >= (long)COUT * 27 * CIN) return;
  const int  cin = (int)(idx % CIN);
  const long t   = idx / CIN;
  const int  tap = (int)(t % 27);
  const long co  = t / 27;
  Wt[idx] = (_Float16)Ws[(co * CIN + cin) * 27 + tap];
}

__global__ __launch_bounds__(256) void prep_lin(const float* __restrict__ lw, _Float16* __restrict__ lwh) {
  const int i = blockIdx.x * 256 + threadIdx.x;
  if (i < 128 * 128) lwh[i] = (_Float16)lw[i];
}

__global__ __launch_bounds__(256) void zero_f32(float* __restrict__ p, int n) {
  const int i = blockIdx.x * 256 + threadIdx.x;
  if (i < n) p[i] = 0.f;
}

// ---------- implicit-GEMM 3x3x3 conv, WMMA f16 -> f32, fused BN-stat atomics ----------
// Block: 256 threads (8 waves), 64M x 128N tile, double-buffered LDS with async G->LDS staging.
// Wave w: M rows [16*(w&3), +16), N cols [64*(w>>2), +64) -> one A frag, 4 B frags, 4 WMMAs/K-step.
// Flattened K loop: kt in [0, 27*CIN/32); tap = kt/(CIN/32), kc = (kt%(CIN/32))*32 (shifts).

template<int CIN, int COUT>
__global__ __launch_bounds__(256) void conv3_wmma(
    const _Float16* __restrict__ X,  // [S_VOX][CIN] f16
    const _Float16* __restrict__ Wt, // [COUT][27][CIN] f16
    float* __restrict__ Y,           // [S_VOX][COUT] f32
    float* __restrict__ csum, float* __restrict__ csq)
{
  constexpr int LD   = 40;            // padded stride (80B rows, 16B-aligned chunks)
  constexpr int KSPT = CIN / 32;      // K-steps per tap: 16 (conv1) / 8 (conv2) — pow2
  constexpr int KT   = 27 * KSPT;
  __shared__ __align__(16) _Float16 sA[2][64  * LD];
  __shared__ __align__(16) _Float16 sB[2][128 * LD];

  const int tid  = threadIdx.x;
  const int lane = tid & 31, wave = tid >> 5;
  const int m0 = blockIdx.x * 64, n0 = blockIdx.y * 128;
  const int wm = (wave & 3) * 16, wn = (wave >> 2) * 64;
  const int ra = tid >> 2, k8 = (tid & 3) * 8;   // staging: row 0..63, 8-half chunk

  // spatial decode of this staging thread's M row
  const int m  = m0 + ra;
  const int l  = m & 15;
  const int hw = m >> 4;
  const int w  = hw % 96, h = hw / 96;

  auto stage = [&](int kt, int buf) {
    const int tap = kt / KSPT;
    const int kc  = (kt % KSPT) * 32;
    const int dl = tap % 3 - 1, dwi = (tap / 3) % 3 - 1, dhi = tap / 9 - 1;
    const int hh2 = h + dhi, ww2 = w + dwi, ll2 = l + dl;
    const bool inb = (unsigned)hh2 < 96u && (unsigned)ww2 < 96u && (unsigned)ll2 < 16u;
    if (inb) {
      const long soff = (long)((hh2 * 96 + ww2) * 16 + ll2) * CIN;
      cp_g2l_b128(X + soff + kc + k8, &sA[buf][ra * LD + k8]);
    } else {
      v8h z = {};
      *(v8h*)&sA[buf][ra * LD + k8] = z;        // DScnt-ordered zero fill for halo
    }
    const _Float16* w0 = Wt + ((long)(n0 + ra)      * 27 + tap) * CIN + kc + k8;
    const _Float16* w1 = Wt + ((long)(n0 + 64 + ra) * 27 + tap) * CIN + kc + k8;
    cp_g2l_b128(w0, &sB[buf][ra * LD + k8]);
    cp_g2l_b128(w1, &sB[buf][(64 + ra) * LD + k8]);
  };

  v8f acc0 = {}, acc1 = {}, acc2 = {}, acc3 = {};

  stage(0, 0);
  async_fence();
  __syncthreads();

  for (int kt = 0; kt < KT; ++kt) {
    const int cur = kt & 1;
    v16h a  = load_a_frag(&sA[cur][wm * LD], LD);
    v16h b0 = load_b_frag(&sB[cur][(wn +  0) * LD], LD);
    v16h b1 = load_b_frag(&sB[cur][(wn + 16) * LD], LD);
    v16h b2 = load_b_frag(&sB[cur][(wn + 32) * LD], LD);
    v16h b3 = load_b_frag(&sB[cur][(wn + 48) * LD], LD);
    if (kt + 1 < KT) stage(kt + 1, cur ^ 1);   // overlap next-step staging with WMMAs
    acc0 = wmma_f16(a, b0, acc0);
    acc1 = wmma_f16(a, b1, acc1);
    acc2 = wmma_f16(a, b2, acc2);
    acc3 = wmma_f16(a, b3, acc3);
    async_fence();
    __syncthreads();
  }

  // epilogue: D layout — lane n=lane&15, VGPR r -> M = r + 8*(lane>>4)
  const int crow = (lane >> 4) * 8;
  const int ccol = lane & 15;
  float s[4] = {0.f, 0.f, 0.f, 0.f};
  float q[4] = {0.f, 0.f, 0.f, 0.f};
  for (int r = 0; r < 8; ++r) {
    const long mg = (long)(m0 + wm + crow + r);
    const float v0 = acc0[r], v1 = acc1[r], v2 = acc2[r], v3 = acc3[r];
    float* yrow = Y + mg * COUT + (n0 + wn + ccol);
    yrow[0]  = v0;  yrow[16] = v1;  yrow[32] = v2;  yrow[48] = v3;
    s[0] += v0; q[0] += v0 * v0;
    s[1] += v1; q[1] += v1 * v1;
    s[2] += v2; q[2] += v2 * v2;
    s[3] += v3; q[3] += v3 * v3;
  }
  for (int t = 0; t < 4; ++t) {
    atomicAdd(&csum[n0 + wn + 16 * t + ccol], s[t]);
    atomicAdd(&csq [n0 + wn + 16 * t + ccol], q[t]);
  }
}

// ---------- BN + ReLU passes ----------

template<int CH>
__global__ __launch_bounds__(256) void bn_relu_half(
    const float* __restrict__ Yin, const float* __restrict__ csum, const float* __restrict__ csq,
    const float* __restrict__ g, const float* __restrict__ b, _Float16* __restrict__ Xout)
{
  const long idx = (long)blockIdx.x * 256 + threadIdx.x;  // over S_VOX*CH, c fastest
  if (idx >= (long)S_VOX * CH) return;
  const int c = (int)(idx % CH);
  const float invn = 1.f / (float)S_VOX;
  const float mean = csum[c] * invn;
  const float var  = csq[c] * invn - mean * mean;
  const float xh   = (Yin[idx] - mean) * rsqrtf(var + 1e-3f);
  Xout[idx] = (_Float16)fmaxf(xh * g[c] + b[c], 0.f);
}

template<int CH>
__global__ __launch_bounds__(256) void bn_relu_out(
    const float* __restrict__ Yin, const float* __restrict__ csum, const float* __restrict__ csq,
    const float* __restrict__ g, const float* __restrict__ b, float* __restrict__ out)
{
  const long idx = (long)blockIdx.x * 256 + threadIdx.x;  // over CH*S_VOX, s fastest (NCHWL out)
  if (idx >= (long)S_VOX * CH) return;
  const long s = idx % S_VOX;
  const int  c = (int)(idx / S_VOX);
  const float invn = 1.f / (float)S_VOX;
  const float mean = csum[c] * invn;
  const float var  = csq[c] * invn - mean * mean;
  const float v    = Yin[s * CH + c];
  out[idx] = fmaxf((v - mean) * rsqrtf(var + 1e-3f) * g[c] + b[c], 0.f);
}

// ---------- host orchestration ----------

extern "C" void kernel_launch(void* const* d_in, const int* in_sizes, int n_in,
                              void* d_out, int out_size, void* d_ws, size_t ws_size,
                              hipStream_t stream)
{
  const float* img   = (const float*)d_in[0];
  const float* pts   = (const float*)d_in[1];
  const float* w1    = (const float*)d_in[2];
  const float* g1    = (const float*)d_in[3];
  const float* b1    = (const float*)d_in[4];
  const float* w2    = (const float*)d_in[5];
  const float* g2    = (const float*)d_in[6];
  const float* b2    = (const float*)d_in[7];
  const float* lw    = (const float*)d_in[8];
  const float* lb    = (const float*)d_in[9];
  const int*   i_img = (const int*)d_in[10];
  const int*   i_pts = (const int*)d_in[11];

  char* p = (char*)d_ws;
  auto take = [&](size_t bytes) { char* r = p; p += (bytes + 255) & ~(size_t)255; return (void*)r; };
  _Float16* X1  = (_Float16*)take((size_t)S_VOX * 512 * 2);
  _Float16* Wt1 = (_Float16*)take((size_t)256 * 27 * 512 * 2);
  float*    Y1  = (float*)   take((size_t)S_VOX * 256 * 4);
  _Float16* X2  = (_Float16*)take((size_t)S_VOX * 256 * 2);
  _Float16* Wt2 = (_Float16*)take((size_t)128 * 27 * 256 * 2);
  float*    Y2  = (float*)   take((size_t)S_VOX * 128 * 4);
  _Float16* lwh = (_Float16*)take((size_t)128 * 128 * 2);
  int* nn_img   = (int*)take(NPT * 4);
  int* nn_pts   = (int*)take(NPT * 4);
  float* stats  = (float*)take(768 * 4);
  float *sum1 = stats, *sq1 = stats + 256, *sum2 = stats + 512, *sq2 = stats + 640;

  zero_f32<<<3, 256, 0, stream>>>(stats, 768);
  prep_conv_w<512,256><<<(256*27*512 + 255)/256, 256, 0, stream>>>(w1, Wt1);
  prep_conv_w<256,128><<<(128*27*256 + 255)/256, 256, 0, stream>>>(w2, Wt2);
  prep_lin<<<64, 256, 0, stream>>>(lw, lwh);
  pack_inputs<<<(S_VOX * 128) / 256, 256, 0, stream>>>(img, pts, X1);

  knn_kernel<<<NPT, 256, 0, stream>>>(i_pts, i_img, NPT, nn_img);
  knn_kernel<<<NPT, 256, 0, stream>>>(i_img, i_pts, NPT, nn_pts);
  fuse_knn_linear<<<NPT/16, 256, 0, stream>>>(img, pts, i_img, i_pts, nn_img, lwh, lb, X1, 256);
  fuse_knn_linear<<<NPT/16, 256, 0, stream>>>(pts, img, i_pts, i_img, nn_pts, lwh, lb, X1, 384);

  conv3_wmma<512,256><<<dim3(S_VOX/64, 2), 256, 0, stream>>>(X1, Wt1, Y1, sum1, sq1);
  bn_relu_half<256><<<(S_VOX * 256) / 256, 256, 0, stream>>>(Y1, sum1, sq1, g1, b1, X2);
  conv3_wmma<256,128><<<dim3(S_VOX/64, 1), 256, 0, stream>>>(X2, Wt2, Y2, sum2, sq2);
  bn_relu_out<128><<<(S_VOX * 128) / 256, 256, 0, stream>>>(Y2, sum2, sq2, g2, b2, (float*)d_out);
}
